// MinkowskiFeatureExtractor_57698590654735
// MI455X (gfx1250) — matrix operand order repaired and checked by reading, not compiled, not run on hardware
//
#include <hip/hip_runtime.h>
#include <hip/hip_bf16.h>

// ---------------------------------------------------------------------------
// CDNA5 (gfx1250) types
// ---------------------------------------------------------------------------
typedef __attribute__((ext_vector_type(8)))  float        v8f;
typedef __attribute__((ext_vector_type(16))) __bf16       v16bf;
typedef __attribute__((ext_vector_type(8)))  unsigned int v8u;

__device__ __forceinline__ unsigned short bfbits(float f) {
    unsigned u = __builtin_bit_cast(unsigned, f);
    return (unsigned short)((u + 0x7FFFu + ((u >> 16) & 1u)) >> 16);
}
__device__ __forceinline__ unsigned packbf2(float lo, float hi) {
    return (unsigned)bfbits(lo) | ((unsigned)bfbits(hi) << 16);
}

// ---------------------------------------------------------------------------
// Sparse gather-conv / dense GEMM via V_WMMA_F32_16X16X32_BF16.
//   Y[Nout,Cout] = sum_t gather(X, kmap[t]) @ W[t]   (idx<0 => zero row)
// kmap==nullptr => identity gather (dense matmul, taps==1).
//
// Block: 128 threads = 4 waves. Tile: 32 rows x 64 cols; wave w owns cols
// w*16..w*16+15 with TWO accumulators (rows 0-15 and 16-31).
// Per (tap, K-chunk of 32) the block stages in LDS:
//   sA : 32 x 32 bf16, row stride 80B  (1 ds_store_b128 / thread)
//   sBt: 64 x 32 bf16, col-major, col stride 80B (8 ds_store_b32 / thread,
//        K-pairs packed into dwords)
// Fragment reads are ds_load_b128 pairs matching ISA 7.12.2 layouts.
// ---------------------------------------------------------------------------
__global__ void __launch_bounds__(128)
sparse_conv_wmma(const float* __restrict__ X,
                 const float* __restrict__ W,
                 const int*   __restrict__ kmap,
                 float* __restrict__ Y,
                 int Nout, int Cin, int Cout, int taps, int fuse_relu) {
    __shared__ __bf16 sA[32 * 40];   // 2560 B
    __shared__ __bf16 sBt[64 * 40];  // 5120 B

    const int tid  = threadIdx.x;
    const int lane = tid & 31;
    const int wav  = tid >> 5;         // 0..3  -> column sub-tile
    const int half = lane >> 4;        // 0/1
    const int m16  = lane & 15;
    const int row0 = blockIdx.x * 32;
    const int colB = blockIdx.y * 64;  // block's first output column

    // cooperative-fill coordinates
    const int arow = tid >> 2;         // 0..31  A row this thread stages
    const int akq  = (tid & 3) * 8;    // first of 8 K this thread stages
    const int bkp  = tid >> 3;         // 0..15  B K-pair (K = 2p, 2p+1)
    const int bcg  = (tid & 7) * 8;    // first of 8 cols this thread stages

    v8f acc0 = {};
    v8f acc1 = {};

    for (int t = 0; t < taps; ++t) {
        if (t + 1 < taps)
            __builtin_prefetch(W + (long long)(t + 1) * Cin * Cout, 0, 1);

        // gather index for the A row this thread stages
        int gidx;
        const int gr = row0 + arow;
        if (kmap) gidx = (gr < Nout) ? kmap[(long long)t * Nout + gr] : -1;
        else      gidx = (gr < Nout) ? gr : -1;
        const float ascale = (gidx >= 0) ? 1.0f : 0.0f;
        const float* xr = X + (long long)(gidx < 0 ? 0 : gidx) * Cin;
        const float* wt = W + (long long)t * Cin * Cout;

        auto do_chunk = [&](int c0) {
            // ---- stage A: 8 channels per thread, one b128 LDS store ----
            float av[8];
            const int cb = c0 + akq;
            if (cb + 7 < Cin) {
                const float4 f0 = *(const float4*)(xr + cb);
                const float4 f1 = *(const float4*)(xr + cb + 4);
                av[0] = f0.x; av[1] = f0.y; av[2] = f0.z; av[3] = f0.w;
                av[4] = f1.x; av[5] = f1.y; av[6] = f1.z; av[7] = f1.w;
            } else {
#pragma unroll
                for (int j = 0; j < 8; ++j)
                    av[j] = (cb + j < Cin) ? xr[cb + j] : 0.0f;
            }
            uint4 ap;
            ap.x = packbf2(av[0] * ascale, av[1] * ascale);
            ap.y = packbf2(av[2] * ascale, av[3] * ascale);
            ap.z = packbf2(av[4] * ascale, av[5] * ascale);
            ap.w = packbf2(av[6] * ascale, av[7] * ascale);
            *(uint4*)((char*)sA + arow * 80 + akq * 2) = ap;

            // ---- stage B transposed: K-pair x 8 cols, 8 b32 LDS stores ----
            const int k0 = c0 + 2 * bkp;
            const bool kv0 = (k0 + 0) < Cin;
            const bool kv1 = (k0 + 1) < Cin;
            const float* wr0 = wt + (long long)(kv0 ? k0     : 0) * Cout + colB + bcg;
            const float* wr1 = wt + (long long)(kv1 ? k0 + 1 : 0) * Cout + colB + bcg;
            float b0[8], b1[8];
            {
                const float4 p0 = *(const float4*)(wr0);
                const float4 p1 = *(const float4*)(wr0 + 4);
                b0[0] = p0.x; b0[1] = p0.y; b0[2] = p0.z; b0[3] = p0.w;
                b0[4] = p1.x; b0[5] = p1.y; b0[6] = p1.z; b0[7] = p1.w;
                const float4 q0 = *(const float4*)(wr1);
                const float4 q1 = *(const float4*)(wr1 + 4);
                b1[0] = q0.x; b1[1] = q0.y; b1[2] = q0.z; b1[3] = q0.w;
                b1[4] = q1.x; b1[5] = q1.y; b1[6] = q1.z; b1[7] = q1.w;
            }
#pragma unroll
            for (int j = 0; j < 8; ++j) {
                const unsigned pk = packbf2(kv0 ? b0[j] : 0.0f,
                                            kv1 ? b1[j] : 0.0f);
                *(unsigned*)((char*)sBt + (bcg + j) * 80 + bkp * 4) = pk;
            }
            __syncthreads();

            // ---- B fragment: two b128 LDS reads (column of sBt) ----
            const int cbn = wav * 16 + m16;
            const uint4 blo = *(const uint4*)((const char*)sBt + cbn * 80 + 16 * half);
            const uint4 bhi = *(const uint4*)((const char*)sBt + cbn * 80 + 32 + 16 * half);
            v8u bu; bu[0] = blo.x; bu[1] = blo.y; bu[2] = blo.z; bu[3] = blo.w;
                    bu[4] = bhi.x; bu[5] = bhi.y; bu[6] = bhi.z; bu[7] = bhi.w;
            const v16bf b = __builtin_bit_cast(v16bf, bu);

            // ---- A fragments (rows 0-15 and 16-31): 4 b128 LDS reads ----
            {
                const uint4 alo = *(const uint4*)((const char*)sA + m16 * 80 + 16 * half);
                const uint4 ahi = *(const uint4*)((const char*)sA + m16 * 80 + 32 + 16 * half);
                v8u au; au[0] = alo.x; au[1] = alo.y; au[2] = alo.z; au[3] = alo.w;
                        au[4] = ahi.x; au[5] = ahi.y; au[6] = ahi.z; au[7] = ahi.w;
                const v16bf a0 = __builtin_bit_cast(v16bf, au);
                acc0 = __builtin_amdgcn_wmma_f32_16x16x32_bf16(
                    false, a0, false, b, (short)0, acc0, false, false);
            }
            {
                const uint4 alo = *(const uint4*)((const char*)sA + (16 + m16) * 80 + 16 * half);
                const uint4 ahi = *(const uint4*)((const char*)sA + (16 + m16) * 80 + 32 + 16 * half);
                v8u au; au[0] = alo.x; au[1] = alo.y; au[2] = alo.z; au[3] = alo.w;
                        au[4] = ahi.x; au[5] = ahi.y; au[6] = ahi.z; au[7] = ahi.w;
                const v16bf a1 = __builtin_bit_cast(v16bf, au);
                acc1 = __builtin_amdgcn_wmma_f32_16x16x32_bf16(
                    false, a1, false, b, (short)0, acc1, false, false);
            }
            __syncthreads();
        };

        int c0 = 0;
        for (; c0 + 64 <= Cin; c0 += 64) {   // 2x unrolled K loop
            do_chunk(c0);
            do_chunk(c0 + 32);
        }
        if (c0 < Cin) do_chunk(c0);
    }

    // ---- write D ----
    const int n = colB + wav * 16 + m16;
    if (n < Cout) {
#pragma unroll
        for (int r = 0; r < 8; ++r) {
            const int m0 = row0 + r + 8 * half;
            const int m1 = m0 + 16;
            float v0 = acc0[r], v1 = acc1[r];
            if (fuse_relu) { v0 = fmaxf(v0, 0.0f); v1 = fmaxf(v1, 0.0f); }
            if (m0 < Nout) Y[(long long)m0 * Cout + n] = v0;
            if (m1 < Nout) Y[(long long)m1 * Cout + n] = v1;
        }
    }
}

// ---------------------------------------------------------------------------
// Instance norm (batch 1): per-channel mean/var over N rows, optional relu
// ---------------------------------------------------------------------------
__global__ void inorm_kernel(float* __restrict__ X, int N, int C, int relu) {
    const int c = blockIdx.x;
    __shared__ float ssum[256];
    __shared__ float ssq[256];
    float s = 0.0f, q = 0.0f;
    for (int i = threadIdx.x; i < N; i += blockDim.x) {
        const float v = X[(long long)i * C + c];
        s += v; q += v * v;
    }
    ssum[threadIdx.x] = s; ssq[threadIdx.x] = q;
    __syncthreads();
    for (int st = blockDim.x >> 1; st > 0; st >>= 1) {
        if ((int)threadIdx.x < st) {
            ssum[threadIdx.x] += ssum[threadIdx.x + st];
            ssq[threadIdx.x]  += ssq[threadIdx.x + st];
        }
        __syncthreads();
    }
    const float mean = ssum[0] / (float)N;
    const float var  = ssq[0] / (float)N - mean * mean;
    const float inv  = rsqrtf(var + 1e-5f);
    for (int i = threadIdx.x; i < N; i += blockDim.x) {
        float v = (X[(long long)i * C + c] - mean) * inv;
        if (relu) v = fmaxf(v, 0.0f);
        X[(long long)i * C + c] = v;
    }
}

__global__ void add_relu_kernel(float* __restrict__ Y, const float* __restrict__ X,
                                long long n) {
    const long long i = (long long)blockIdx.x * blockDim.x + threadIdx.x;
    if (i < n) Y[i] = fmaxf(Y[i] + X[i], 0.0f);
}

__global__ void concat_kernel(const float* __restrict__ A, int Ca,
                              const float* __restrict__ B, int Cb,
                              float* __restrict__ Y, int N) {
    const int Cd = Ca + Cb;
    const long long total = (long long)N * Cd;
    const long long i = (long long)blockIdx.x * blockDim.x + threadIdx.x;
    if (i >= total) return;
    const long long r = i / Cd;
    const int       c = (int)(i % Cd);
    Y[i] = (c < Ca) ? A[r * Ca + c] : B[r * Cb + (c - Ca)];
}

__global__ void bias_norm_kernel(float* __restrict__ X,
                                 const float* __restrict__ bias, int N, int C) {
    const int i = blockIdx.x * blockDim.x + threadIdx.x;
    if (i >= N) return;
    float s = 0.0f;
    for (int c = 0; c < C; ++c) {
        const float v = X[(long long)i * C + c] + bias[c];
        s += v * v;
    }
    const float inv = rsqrtf(s);
    for (int c = 0; c < C; ++c)
        X[(long long)i * C + c] = (X[(long long)i * C + c] + bias[c]) * inv;
}

// ---------------------------------------------------------------------------
// Brute-force top-k (k<=8) nearest voxels + IDW, voxel list tiled via LDS
// ---------------------------------------------------------------------------
#define UPS_TILE 256
__global__ void __launch_bounds__(128)
upsample_kernel(const float* __restrict__ xyz,
                const float* __restrict__ vxyz,
                const float* __restrict__ feat,
                const int* __restrict__ kptr,
                float* __restrict__ out,
                int Nq, int Nv, int C) {
    __shared__ float sv[UPS_TILE * 3];
    const int i = blockIdx.x * blockDim.x + threadIdx.x;
    const bool active = i < Nq;

    int k = *kptr;
    k = k < 1 ? 1 : (k > 8 ? 8 : k);

    float px = 0.f, py = 0.f, pz = 0.f;
    if (active) { px = xyz[i * 3]; py = xyz[i * 3 + 1]; pz = xyz[i * 3 + 2]; }

    float bd[8]; int bi[8];
#pragma unroll
    for (int j = 0; j < 8; ++j) { bd[j] = 3.0e38f; bi[j] = 0; }

    for (int v0 = 0; v0 < Nv; v0 += UPS_TILE) {
        const int cnt = (Nv - v0 < UPS_TILE) ? (Nv - v0) : UPS_TILE;
        __syncthreads();
        for (int j = threadIdx.x; j < cnt * 3; j += blockDim.x)
            sv[j] = vxyz[(long long)v0 * 3 + j];
        __syncthreads();
        if (active) {
            for (int vv = 0; vv < cnt; ++vv) {
                const float dx = px - sv[vv * 3 + 0];
                const float dy = py - sv[vv * 3 + 1];
                const float dz = pz - sv[vv * 3 + 2];
                const float sq = dx * dx + dy * dy + dz * dz;
                if (sq < bd[k - 1]) {
                    int p = k - 1;
                    while (p > 0 && bd[p - 1] > sq) {
                        bd[p] = bd[p - 1]; bi[p] = bi[p - 1]; --p;
                    }
                    bd[p] = sq; bi[p] = v0 + vv;
                }
            }
        }
    }
    if (!active) return;

    float w[8], wsum = 0.0f;
    for (int j = 0; j < k; ++j) {
        const float d = sqrtf(fmaxf(bd[j], 1e-12f));
        w[j] = 1.0f / (d + 1e-7f);
        wsum += w[j];
    }
    const float winv = 1.0f / wsum;
    for (int c = 0; c < C; ++c) {
        float acc = 0.0f;
        for (int j = 0; j < k; ++j)
            acc += w[j] * feat[(long long)bi[j] * C + c];
        out[(long long)i * C + c] = acc * winv;
    }
}

// ---------------------------------------------------------------------------
// Host orchestration
// ---------------------------------------------------------------------------
extern "C" void kernel_launch(void* const* d_in, const int* in_sizes, int n_in,
                              void* d_out, int out_size, void* d_ws, size_t ws_size,
                              hipStream_t stream) {
    const float* W_conv1   = (const float*)d_in[0];   // [343,3,64]
    const float* W_block1  = (const float*)d_in[1];   // [2,27,64,64]
    const float* W_conv2   = (const float*)d_in[2];   // [27,64,64]
    const float* W_block2  = (const float*)d_in[3];
    const float* W_conv3   = (const float*)d_in[4];   // [27,64,128]
    const float* W_block3  = (const float*)d_in[5];   // [2,27,128,128]
    const float* W_conv4   = (const float*)d_in[6];   // [27,128,128]
    const float* W_block4  = (const float*)d_in[7];
    const float* W_conv4tr = (const float*)d_in[8];   // [27,128,128]
    const float* W_block4tr= (const float*)d_in[9];
    const float* W_conv3tr = (const float*)d_in[10];  // [27,256,128]
    const float* W_block3tr= (const float*)d_in[11];
    const float* W_conv2tr = (const float*)d_in[12];  // [27,192,128]
    const float* W_block2tr= (const float*)d_in[13];
    const float* W_conv1tr = (const float*)d_in[14];  // [192,64]
    const float* W_final   = (const float*)d_in[15];  // [64,64]
    const float* b_final   = (const float*)d_in[16];  // [64]
    const int*   k_values  = (const int*)d_in[45];

    auto conv = [&](const float* X, const float* W, const int* km, float* Y,
                    int Nout, int Cin, int Cout, int taps, int frelu = 0) {
        dim3 grid((Nout + 31) / 32, (Cout + 63) / 64);
        sparse_conv_wmma<<<grid, dim3(128), 0, stream>>>(X, W, km, Y,
                                                         Nout, Cin, Cout, taps, frelu);
    };
    auto inorm = [&](float* X, int N, int C, int relu) {
        inorm_kernel<<<C, 256, 0, stream>>>(X, N, C, relu);
    };
    auto add_relu = [&](float* Y, const float* X, long long n) {
        add_relu_kernel<<<(unsigned)((n + 255) / 256), 256, 0, stream>>>(Y, X, n);
    };
    auto cat = [&](const float* A, int Ca, const float* B, int Cb, float* Y, int N) {
        long long tot = (long long)N * (Ca + Cb);
        concat_kernel<<<(unsigned)((tot + 255) / 256), 256, 0, stream>>>(A, Ca, B, Cb, Y, N);
    };
    auto res_block = [&](float* in, float* t1, float* out, const float* W2,
                         const int* km, int N, int C) {
        conv(in, W2, km, t1, N, C, C, 27);
        inorm(t1, N, C, 1);
        conv(t1, W2 + (long long)27 * C * C, km, out, N, C, C, 27);
        inorm(out, N, C, 0);
        add_relu(out, in, (long long)N * C);
    };

    for (int cloud = 0; cloud < 2; ++cloud) {
        const int base = 17 + cloud * 14;
        const float* feats   = (const float*)d_in[base + 0];
        const float* xyz     = (const float*)d_in[base + 1];
        const float* vcoords = (const float*)d_in[base + 2];
        const int* km7 = (const int*)d_in[base + 3];
        const int* km0 = (const int*)d_in[base + 4];
        const int* km1 = (const int*)d_in[base + 5];
        const int* km2 = (const int*)d_in[base + 6];
        const int* km3 = (const int*)d_in[base + 7];
        const int* kd0 = (const int*)d_in[base + 8];
        const int* ku0 = (const int*)d_in[base + 9];
        const int* kd1 = (const int*)d_in[base + 10];
        const int* ku1 = (const int*)d_in[base + 11];
        const int* kd2 = (const int*)d_in[base + 12];
        const int* ku2 = (const int*)d_in[base + 13];

        const int N0 = in_sizes[base + 0] / 3;      // feats [N0,3]
        const int Nq = in_sizes[base + 1] / 3;      // xyz   [1,Nq,3]
        const int N1 = in_sizes[base + 8] / 27;     // kd0 [27,N1]
        const int N2 = in_sizes[base + 10] / 27;    // kd1 [27,N2]
        const int N3 = in_sizes[base + 12] / 27;    // kd2 [27,N3]

        // ---- bump-allocate fp32 scratch (reset per cloud) ----
        char* wp = (char*)d_ws;
        auto alloc = [&](long long nfloats) {
            float* p = (float*)wp;
            wp += ((nfloats * 4 + 255) / 256) * 256;
            return p;
        };
        float* s1 = alloc((long long)N0 * 64);
        float* s2 = alloc((long long)N1 * 64);
        float* s4 = alloc((long long)N2 * 128);
        float* A  = alloc((long long)N0 * 256);
        float* B  = alloc((long long)N0 * 256);
        float* Cb = alloc((long long)N0 * 256);
        (void)ws_size;

        // ---- encoder ----
        conv(feats, W_conv1, km7, s1, N0, 3, 64, 343);
        inorm(s1, N0, 64, 0);
        res_block(s1, A, B, W_block1, km0, N0, 64);          // cur0 = B

        conv(B, W_conv2, kd0, s2, N1, 64, 64, 27);
        inorm(s2, N1, 64, 0);
        res_block(s2, A, Cb, W_block2, km1, N1, 64);         // cur1 = Cb

        conv(Cb, W_conv3, kd1, s4, N2, 64, 128, 27);
        inorm(s4, N2, 128, 0);
        res_block(s4, A, B, W_block3, km2, N2, 128);         // cur2 = B

        conv(B, W_conv4, kd2, Cb, N3, 128, 128, 27);         // s8 = Cb
        inorm(Cb, N3, 128, 0);
        res_block(Cb, A, B, W_block4, km3, N3, 128);         // cur3 = B

        // ---- decoder ----
        conv(B, W_conv4tr, ku2, Cb, N2, 128, 128, 27);
        inorm(Cb, N2, 128, 0);
        res_block(Cb, A, B, W_block4tr, km2, N2, 128);
        cat(B, 128, s4, 128, Cb, N2);                        // (N2,256)

        conv(Cb, W_conv3tr, ku1, A, N1, 256, 128, 27);
        inorm(A, N1, 128, 0);
        res_block(A, Cb, B, W_block3tr, km1, N1, 128);
        cat(B, 128, s2, 64, Cb, N1);                         // (N1,192)

        conv(Cb, W_conv2tr, ku0, A, N0, 192, 128, 27);
        inorm(A, N0, 128, 0);
        res_block(A, Cb, B, W_block2tr, km0, N0, 128);
        cat(B, 128, s1, 64, Cb, N0);                         // (N0,192)

        // ---- dense head (relu fused into first GEMM epilogue) ----
        conv(Cb, W_conv1tr, nullptr, A, N0, 192, 64, 1, /*relu=*/1);
        conv(A, W_final, nullptr, B, N0, 64, 64, 1);
        bias_norm_kernel<<<(N0 + 127) / 128, 128, 0, stream>>>(B, b_final, N0, 64);

        // ---- kNN IDW upsample ----
        float* outp = (float*)d_out + (long long)cloud * Nq * 64;
        upsample_kernel<<<(Nq + 127) / 128, 128, 0, stream>>>(
            xyz, vcoords, B, k_values, outp, Nq, N0, 64);
    }
    (void)n_in; (void)out_size;
}
